// VQVAEAuxiliary_28716151341699
// MI455X (gfx1250) — compile-verified, matrix operand order checked by reading, not run on hardware
//
#include <hip/hip_runtime.h>
#include <hip/hip_bf16.h>
#include <math.h>

// ---------------- types for WMMA / TDM (gfx1250, wave32) ----------------
typedef __attribute__((ext_vector_type(16))) __bf16 v16bf;
typedef __attribute__((ext_vector_type(2)))  __bf16 bf16x2;
typedef __attribute__((ext_vector_type(8)))  float  v8f;
typedef __attribute__((ext_vector_type(4)))  unsigned int u32x4;
typedef __attribute__((ext_vector_type(8)))  int i32x8;
typedef __attribute__((ext_vector_type(4)))  int i32x4;

#define DEV __device__ __forceinline__

static constexpr int NB = 4096;   // batch
static constexpr int ND = 128;    // embedding dim
static constexpr int NK = 8192;   // codebook size
static constexpr int F0 = 64*7*7; // 3136 linear out features

// ---------------- fragment index helpers (ISA 7.12.2, 16-bit A 16x32) -----
// lane: M (or N) = lane&15 ; half = lane>>4
// VGPR v (=j>>1): K = (v&4?16:0) + 2*(v&3) + 8*half + (j&1)
DEV int aK(int j, int hlf) {
  int v = j >> 1;
  return ((v & 4) ? 16 : 0) + 2 * (v & 3) + 8 * hlf + (j & 1);
}

DEV v16bf load_frag_bf16(const __bf16* rowbase, int hlf) {
  v16bf f;
#pragma unroll
  for (int j = 0; j < 16; j += 2) {
    int k = aK(j, hlf);                       // even K -> dword aligned pair
    bf16x2 p = *(const bf16x2*)(rowbase + k);
    f[j] = p[0]; f[j + 1] = p[1];
  }
  return f;
}

DEV v16bf load_frag_f32cvt(const float* rowbase, int hlf) {
  v16bf f;
#pragma unroll
  for (int j = 0; j < 16; ++j) f[j] = (__bf16)rowbase[aK(j, hlf)];
  return f;
}

DEV v8f wmma_bf16(v16bf a, v16bf b, v8f c) {
  return __builtin_amdgcn_wmma_f32_16x16x32_bf16(
      false, a, false, b, (short)0, c, false, false);
}

// ---------------- Tensor Data Mover: flat 1-D copy of nelem bf16 -> LDS ---
#if __has_builtin(__builtin_amdgcn_tensor_load_to_lds)
#define HAVE_TDM 1
#else
#define HAVE_TDM 0
#endif

#if HAVE_TDM
// D# per ISA 8.3/8.4: group0 {count=1, lds_addr, global_addr, type=2},
// group1 {data_size=2B, tensor_dim0=tile_dim0=nelem, tensor_dim1=1}
// Toolchain here uses the 6-arg builtin: (g0, g1, g2, g3, g4, cpol).
DEV void tdm_load_1d(unsigned lds_addr, const void* gptr, unsigned nelem) {
  unsigned long long ga = (unsigned long long)(size_t)gptr;
  u32x4 g0;
  g0[0] = 1u;                                            // count=1
  g0[1] = lds_addr;                                      // bits 63:32
  g0[2] = (unsigned)ga;                                  // bits 95:64
  g0[3] = (unsigned)((ga >> 32) & 0x01FFFFFFu) | (2u << 30); // addr hi, type=2
  i32x8 g1;
  g1[0] = (int)(1u << 16);                               // data_size=1 (2 bytes)
  g1[1] = (int)((nelem & 0xFFFFu) << 16);                // tensor_dim0[15:0]
  g1[2] = (int)(((nelem >> 16) & 0xFFFFu) | (1u << 16)); // td0[31:16], td1=1
  g1[3] = (int)((nelem & 0xFFFFu) << 16);                // tile_dim0 = nelem
  g1[4] = 0;                                             // tile_dim1/2 unused
  g1[5] = (int)nelem;                                    // td0_stride lo
  g1[6] = 0;
  g1[7] = 0;
  i32x4 z4 = {0, 0, 0, 0};
  i32x8 z8 = {0, 0, 0, 0, 0, 0, 0, 0};
  __builtin_amdgcn_tensor_load_to_lds(g0, g1, z4, z4, z8, 0);
}
#endif

// ---------------- small prep kernels ----------------
__global__ void k_esq(const float* __restrict__ cb, float* __restrict__ esq) {
  int k = blockIdx.x * blockDim.x + threadIdx.x;
  if (k >= NK) return;
  const float* r = cb + (size_t)k * ND;
  float s = 0.f;
#pragma unroll 8
  for (int d = 0; d < ND; ++d) s += r[d] * r[d];
  esq[k] = s;
}

__global__ void k_cvt_bf16(const float* __restrict__ src, __bf16* __restrict__ dst, long n) {
  long i = (long)blockIdx.x * blockDim.x + threadIdx.x;
  if (i < n) dst[i] = (__bf16)src[i];
}

// w: (Cin, Cout, 4, 4) f32  ->  wT: [tap=ky*4+kx][cout][cin] bf16
__global__ void k_wT(const float* __restrict__ w, __bf16* __restrict__ wT, int Cin, int Cout) {
  int i = blockIdx.x * blockDim.x + threadIdx.x;
  int total = Cin * Cout * 16;
  if (i >= total) return;
  int ci  = i / (Cout * 16);
  int rem = i % (Cout * 16);
  int co  = rem / 16;
  int t   = rem % 16;
  wT[((size_t)t * Cout + co) * Cin + ci] = (__bf16)w[i];
}

// ---------------- VQ: fused distance GEMM + argmin ----------------
__global__ __launch_bounds__(128) void k_vq(const float* __restrict__ z,
                                            const __bf16* __restrict__ cbbf,
                                            const float* __restrict__ esq,
                                            int* __restrict__ idx_out,
                                            float* __restrict__ idx_f_out) {
  __shared__ float s_zsq[16];
  __shared__ float s_wmin[4][16];
  __shared__ int   s_widx[4][16];
  int tid = threadIdx.x, wave = tid >> 5, lane = tid & 31;
  int hlf = lane >> 4, ln = lane & 15;
  int row0 = blockIdx.x * 16;

  if (tid < 16) {
    const float* zr = z + (size_t)(row0 + tid) * ND;
    float s = 0.f;
#pragma unroll 8
    for (int d = 0; d < ND; ++d) s += zr[d] * zr[d];
    s_zsq[tid] = s;
  }
  __syncthreads();

  v16bf a[4];
  const float* zrow = z + (size_t)(row0 + ln) * ND;
#pragma unroll
  for (int s = 0; s < 4; ++s) a[s] = load_frag_f32cvt(zrow + 32 * s, hlf);

  float rmin[8]; int ridx[8]; float myzsq[8];
#pragma unroll
  for (int r = 0; r < 8; ++r) { rmin[r] = 3.0e38f; ridx[r] = 0x7fffffff; myzsq[r] = s_zsq[r + 8 * hlf]; }

  for (int ct = wave; ct < NK / 16; ct += 4) {
    int col0 = ct * 16;
    v8f c = {0.f, 0.f, 0.f, 0.f, 0.f, 0.f, 0.f, 0.f};
    const __bf16* brow = cbbf + (size_t)(col0 + ln) * ND;
    // prefetch this wave's next code tile (64 codes ahead) while WMMAs run
    __builtin_prefetch((const void*)(brow + (size_t)64 * ND), 0, 3);
#pragma unroll
    for (int s = 0; s < 4; ++s) {
      v16bf b = load_frag_bf16(brow + 32 * s, hlf);
      c = wmma_bf16(a[s], b, c);
    }
    float e = esq[col0 + ln];
    int id = col0 + ln;
#pragma unroll
    for (int r = 0; r < 8; ++r) {
      float dist = myzsq[r] + e - 2.0f * c[r];
      if (dist < rmin[r] || (dist == rmin[r] && id < ridx[r])) { rmin[r] = dist; ridx[r] = id; }
    }
  }
#pragma unroll
  for (int m = 1; m < 16; m <<= 1) {
#pragma unroll
    for (int r = 0; r < 8; ++r) {
      float ov = __shfl_xor(rmin[r], m, 32);
      int   oi = __shfl_xor(ridx[r], m, 32);
      if (ov < rmin[r] || (ov == rmin[r] && oi < ridx[r])) { rmin[r] = ov; ridx[r] = oi; }
    }
  }
  if (ln == 0) {
    int ro = 8 * hlf;
#pragma unroll
    for (int r = 0; r < 8; ++r) { s_wmin[wave][ro + r] = rmin[r]; s_widx[wave][ro + r] = ridx[r]; }
  }
  __syncthreads();
  if (tid < 16) {
    float bv = s_wmin[0][tid]; int bi = s_widx[0][tid];
#pragma unroll
    for (int w = 1; w < 4; ++w) {
      float v = s_wmin[w][tid]; int i2 = s_widx[w][tid];
      if (v < bv || (v == bv && i2 < bi)) { bv = v; bi = i2; }
    }
    idx_out[row0 + tid] = bi;
    idx_f_out[row0 + tid] = (float)bi;
  }
}

// ---------------- gather z_q, per-row loss partials ----------------
__global__ __launch_bounds__(128) void k_gather(const float* __restrict__ z,
                                                const float* __restrict__ cb,
                                                const int* __restrict__ idx,
                                                __bf16* __restrict__ zqbf,
                                                float* __restrict__ partial) {
  int b = blockIdx.x, t = threadIdx.x;     // 128 threads == ND
  int id = idx[b];
  float zq = cb[(size_t)id * ND + t];
  float zv = z[(size_t)b * ND + t];
  float d = zv - zq;
  zqbf[(size_t)b * ND + t] = (__bf16)zq;
  __shared__ float red[128];
  red[t] = d * d; __syncthreads();
  for (int s = 64; s > 0; s >>= 1) { if (t < s) red[t] += red[t + s]; __syncthreads(); }
  if (t == 0) partial[b] = red[0];
}

__global__ void k_loss(const float* __restrict__ partial, float* __restrict__ out_losses) {
  __shared__ float red[256];
  float s = 0.f;
  for (int i = threadIdx.x; i < NB; i += 256) s += partial[i];
  red[threadIdx.x] = s; __syncthreads();
  for (int st = 128; st > 0; st >>= 1) { if (threadIdx.x < st) red[threadIdx.x] += red[threadIdx.x + st]; __syncthreads(); }
  if (threadIdx.x == 0) {
    float m = red[0] / (float)((size_t)NB * ND);
    out_losses[0] = m;
    out_losses[1] = m;
  }
}

// ---------------- Linear (WMMA), output NHWC (B,7,7,64) bf16 ----------------
__global__ __launch_bounds__(32) void k_linear(const __bf16* __restrict__ A,
                                               const __bf16* __restrict__ Wb,
                                               const float* __restrict__ bias,
                                               __bf16* __restrict__ x0) {
  int n0 = blockIdx.x * 16, m0 = blockIdx.y * 16;
  int lane = threadIdx.x, hlf = lane >> 4, ln = lane & 15;
  v16bf a[4];
  const __bf16* arow = A + (size_t)(m0 + ln) * ND;
#pragma unroll
  for (int s = 0; s < 4; ++s) a[s] = load_frag_bf16(arow + 32 * s, hlf);
  v8f c = {0.f, 0.f, 0.f, 0.f, 0.f, 0.f, 0.f, 0.f};
  const __bf16* brow = Wb + (size_t)(n0 + ln) * ND;
#pragma unroll
  for (int s = 0; s < 4; ++s) {
    v16bf b = load_frag_bf16(brow + 32 * s, hlf);
    c = wmma_bf16(a[s], b, c);
  }
  int n = n0 + ln;
  int cch = n / 49, p = n % 49;            // feature n = c*49 + (iy*7+ix)
  float bv = bias[n];
#pragma unroll
  for (int r = 0; r < 8; ++r) {
    int bi = m0 + r + 8 * hlf;
    x0[((size_t)bi * 49 + p) * 64 + cch] = (__bf16)(c[r] + bv);
  }
}

// ---------------- ConvTranspose2d k4 s2 p1 via per-tap WMMA GEMMs ----------
// 4-wave block: 64 batches at one output pixel. Tap weight tiles staged in
// LDS once per block (TDM when available), B-fragments read via ds_load.
template <int Cin, int Cout, int Hin, int Win, bool RELU>
__global__ __launch_bounds__(128) void k_convT(const __bf16* __restrict__ xin,
                                               const __bf16* __restrict__ wT,
                                               const float* __restrict__ bias,
                                               __bf16* __restrict__ xout) {
  constexpr int Hout = 2 * Hin, Wout = 2 * Win;
  constexpr int NT = Cout / 16, KS = Cin / 32;
  constexpr int TAPW = Cout * Cin;
  __shared__ __bf16 sw[4 * TAPW];

  int pix = blockIdx.x;
  int oy = pix / Wout, ox = pix % Wout;
  int tid = threadIdx.x, wave = tid >> 5, lane = tid & 31;
  int hlf = lane >> 4, ln = lane & 15;
  int bw = blockIdx.y * 64 + wave * 16;

  // enumerate valid taps (uniform across block: depends only on blockIdx)
  int tap[4], tiy[4], tix[4], ntap = 0;
  for (int ky = 0; ky < 4; ++ky) {
    int ty = oy + 1 - ky; if (ty < 0 || (ty & 1)) continue;
    int iy = ty >> 1; if (iy >= Hin) continue;
    for (int kx = 0; kx < 4; ++kx) {
      int tx = ox + 1 - kx; if (tx < 0 || (tx & 1)) continue;
      int ix = tx >> 1; if (ix >= Win) continue;
      tap[ntap] = ky * 4 + kx; tiy[ntap] = iy; tix[ntap] = ix; ++ntap;
    }
  }

  // stage tap weight tiles into LDS
#if HAVE_TDM
  if (tid < 32) {                      // wave 0 issues the DMAs
    for (int t = 0; t < ntap; ++t)
      tdm_load_1d((unsigned)(size_t)&sw[t * TAPW],
                  wT + (size_t)tap[t] * TAPW, TAPW);
    __builtin_amdgcn_s_wait_tensorcnt(0);
  }
#else
  for (int t = 0; t < ntap; ++t) {
    const __bf16* src = wT + (size_t)tap[t] * TAPW;
    for (int i = tid; i < TAPW; i += 128) sw[t * TAPW + i] = src[i];
  }
#endif
  __syncthreads();

  v8f acc[NT];
#pragma unroll
  for (int nt = 0; nt < NT; ++nt) acc[nt] = (v8f){0.f,0.f,0.f,0.f,0.f,0.f,0.f,0.f};

  for (int t = 0; t < ntap; ++t) {
    const __bf16* arow = xin + (((size_t)(bw + ln) * Hin + tiy[t]) * Win + tix[t]) * Cin;
    const __bf16* wl = sw + t * TAPW;
#pragma unroll
    for (int s = 0; s < KS; ++s) {
      v16bf a = load_frag_bf16(arow + 32 * s, hlf);
#pragma unroll
      for (int nt = 0; nt < NT; ++nt) {
        v16bf b = load_frag_bf16(wl + (size_t)(nt * 16 + ln) * Cin + 32 * s, hlf);
        acc[nt] = wmma_bf16(a, b, acc[nt]);
      }
    }
  }
#pragma unroll
  for (int nt = 0; nt < NT; ++nt) {
    int co = nt * 16 + ln;
    float bv = bias[co];
#pragma unroll
    for (int r = 0; r < 8; ++r) {
      int bi = bw + r + 8 * hlf;
      float v = acc[nt][r] + bv;
      if (RELU) v = fmaxf(v, 0.0f);
      xout[(((size_t)bi * Hout + oy) * Wout + ox) * Cout + co] = (__bf16)v;
    }
  }
}

// ---------------- final convT 32->1 (scalar, small) ----------------
__global__ void k_conv3(const __bf16* __restrict__ x2, const float* __restrict__ w3,
                        const float* __restrict__ b3, float* __restrict__ x3) {
  int i = blockIdx.x * 256 + threadIdx.x;
  if (i >= NB * 56 * 56) return;
  int b = i / (56 * 56), p = i % (56 * 56);
  int oy = p / 56, ox = p % 56;
  float acc = b3[0];
#pragma unroll
  for (int ky = 0; ky < 4; ++ky) {
    int ty = oy + 1 - ky; if (ty < 0 || (ty & 1)) continue;
    int iy = ty >> 1; if (iy >= 28) continue;
#pragma unroll
    for (int kx = 0; kx < 4; ++kx) {
      int tx = ox + 1 - kx; if (tx < 0 || (tx & 1)) continue;
      int ix = tx >> 1; if (ix >= 28) continue;
      const __bf16* xr = x2 + (((size_t)b * 28 + iy) * 28 + ix) * 32;
#pragma unroll
      for (int ci = 0; ci < 32; ++ci)
        acc += (float)xr[ci] * w3[ci * 16 + ky * 4 + kx];
    }
  }
  x3[i] = acc;
}

// ---------------- bilinear 56->84 (half-pixel) + tanh ----------------
__global__ void k_resize(const float* __restrict__ x3, float* __restrict__ out) {
  int i = blockIdx.x * 256 + threadIdx.x;
  if (i >= NB * 84 * 84) return;
  int b = i / 7056, p = i % 7056;
  int oy = p / 84, ox = p % 84;
  const float sc = 56.0f / 84.0f;
  float sy = fminf(fmaxf((oy + 0.5f) * sc - 0.5f, 0.0f), 55.0f);
  float sx = fminf(fmaxf((ox + 0.5f) * sc - 0.5f, 0.0f), 55.0f);
  int y0 = (int)sy, x0 = (int)sx;
  int y1 = y0 + 1 > 55 ? 55 : y0 + 1;
  int x1 = x0 + 1 > 55 ? 55 : x0 + 1;
  float fy = sy - y0, fx = sx - x0;
  const float* base = x3 + (size_t)b * 3136;
  float v00 = base[y0 * 56 + x0], v01 = base[y0 * 56 + x1];
  float v10 = base[y1 * 56 + x0], v11 = base[y1 * 56 + x1];
  float v = v00 * (1 - fy) * (1 - fx) + v01 * (1 - fy) * fx
          + v10 * fy * (1 - fx) + v11 * fy * fx;
  out[i] = tanhf(v);
}

// ---------------- workspace layout (bytes) ----------------
static constexpr size_t OFF_ESQ   = 0;
static constexpr size_t OFF_IDX   = 32768;
static constexpr size_t OFF_PART  = 49152;
static constexpr size_t OFF_ZQBF  = 65536;
static constexpr size_t OFF_LINW  = 1114112;
static constexpr size_t OFF_CBBF  = 1916928;
static constexpr size_t OFF_W1T   = 4014080;
static constexpr size_t OFF_W2T   = 4145152;
static constexpr size_t OFF_X0    = 4210688;
static constexpr size_t OFF_X1    = 29900800;
static constexpr size_t OFF_X2    = 132661248;
static constexpr size_t OFF_X3    = 338182144;

extern "C" void kernel_launch(void* const* d_in, const int* in_sizes, int n_in,
                              void* d_out, int out_size, void* d_ws, size_t ws_size,
                              hipStream_t stream) {
  const float* z     = (const float*)d_in[0];
  const float* cb    = (const float*)d_in[1];
  const float* lin_w = (const float*)d_in[2];
  const float* lin_b = (const float*)d_in[3];
  const float* w1    = (const float*)d_in[4];
  const float* b1    = (const float*)d_in[5];
  const float* w2    = (const float*)d_in[6];
  const float* b2    = (const float*)d_in[7];
  const float* w3    = (const float*)d_in[8];
  const float* b3    = (const float*)d_in[9];

  char* ws = (char*)d_ws;
  float*  esq   = (float*)(ws + OFF_ESQ);
  int*    idx   = (int*)(ws + OFF_IDX);
  float*  part  = (float*)(ws + OFF_PART);
  __bf16* zqbf  = (__bf16*)(ws + OFF_ZQBF);
  __bf16* linwb = (__bf16*)(ws + OFF_LINW);
  __bf16* cbbf  = (__bf16*)(ws + OFF_CBBF);
  __bf16* w1T   = (__bf16*)(ws + OFF_W1T);
  __bf16* w2T   = (__bf16*)(ws + OFF_W2T);
  __bf16* x0    = (__bf16*)(ws + OFF_X0);
  __bf16* x1    = (__bf16*)(ws + OFF_X1);
  __bf16* x2    = (__bf16*)(ws + OFF_X2);
  float*  x3    = (float*)(ws + OFF_X3);

  float* out    = (float*)d_out;
  float* idx_f  = out + (size_t)NB * 84 * 84;   // 28901376
  float* losses = idx_f + NB;                   // 28905472

  k_esq<<<NK / 256, 256, 0, stream>>>(cb, esq);
  k_cvt_bf16<<<(NK * ND + 255) / 256, 256, 0, stream>>>(cb, cbbf, (long)NK * ND);
  k_cvt_bf16<<<(F0 * ND + 255) / 256, 256, 0, stream>>>(lin_w, linwb, (long)F0 * ND);
  k_wT<<<(64 * 64 * 16) / 256, 256, 0, stream>>>(w1, w1T, 64, 64);
  k_wT<<<(64 * 32 * 16) / 256, 256, 0, stream>>>(w2, w2T, 64, 32);

  k_vq<<<NB / 16, 128, 0, stream>>>(z, cbbf, esq, idx, idx_f);

  k_gather<<<NB, 128, 0, stream>>>(z, cb, idx, zqbf, part);
  k_loss<<<1, 256, 0, stream>>>(part, losses);

  k_linear<<<dim3(F0 / 16, NB / 16), 32, 0, stream>>>(zqbf, linwb, lin_b, x0);

  k_convT<64, 64, 7, 7, true><<<dim3(14 * 14, NB / 64), 128, 0, stream>>>(x0, w1T, b1, x1);
  k_convT<64, 32, 14, 14, true><<<dim3(28 * 28, NB / 64), 128, 0, stream>>>(x1, w2T, b2, x2);

  k_conv3<<<(NB * 56 * 56 + 255) / 256, 256, 0, stream>>>(x2, w3, b3, x3);
  k_resize<<<(NB * 84 * 84 + 255) / 256, 256, 0, stream>>>(x3, out);
}
